// QuantumLayer_50964081934821
// MI455X (gfx1250) — matrix operand order repaired and checked by reading, not compiled, not run on hardware
//
#include <hip/hip_runtime.h>

// ---------------------------------------------------------------------------
// QuantumLayer on MI455X (gfx1250, wave32, WMMA)
//
//   out[b,o] = sum_k x[b,k]*W[o,k] + bias[o] + mean(q[b,:])/15
//
// M=8192 N=4096 K=4096 fp32 GEMM (275 GFLOP): compute-bound (HBM floor ~14us
// vs ~110us bf16-WMMA compute), so precision path dominates. We use
// split-precision bf16x3 (x=xh+xl, W=wh+wl, acc += xh*wh + xh*wl + xl*wh,
// f32 accumulate) -- near-fp32 accuracy at 3x bf16 cost, ~2.7x cheaper than
// the K=4 fp32 WMMA path.
//
// v2 changes (driven by round-1 histogram valu=622 / wmma=24):
//  * fp32->bf16 hi/lo split now happens ONCE per block at LDS staging time
//    into four bf16 planes (xh/xl/wh/wl) -- kills the per-wave re-conversion
//    and the v_mov_b16 fragment assembly that dominated the VALU stream.
//  * Fragment builds are pure ds_load_b128 + shufflevector (zero ALU).
//  * Double-buffered LDS pipeline: global loads for chunk i+1 issue before
//    the 24 WMMAs of chunk i; converted planes commit to the alternate
//    buffer; ONE s_barrier per K-chunk.
// ---------------------------------------------------------------------------

typedef __attribute__((ext_vector_type(16))) __bf16 bf16x16;
typedef __attribute__((ext_vector_type(8)))  __bf16 bf16x8;
typedef __attribute__((ext_vector_type(4)))  __bf16 bf16x4;
typedef __attribute__((ext_vector_type(8)))  float  floatx8;

#define BATCH  8192
#define IN_K   4096
#define OUT_N  4096
#define SHOTS  1000

#define BM 128           // block tile M
#define BN 128           // block tile N
#define BK 32            // K chunk (one bf16 WMMA K-step)
#define LDSH (BK + 8)    // bf16 row stride: 40 bf16 = 80B (16B-aligned, bank-spread)

// ---------------------------------------------------------------------------
// Quantum expectation: e[b] = (sum_s q[b,s]) / (SHOTS * 15).  One wave per row.
// ---------------------------------------------------------------------------
__global__ __launch_bounds__(256) void qexp_kernel(const int* __restrict__ q,
                                                   float* __restrict__ e) {
    const int wave = threadIdx.x >> 5;   // 8 waves / block (wave32)
    const int lane = threadIdx.x & 31;
    const int row  = blockIdx.x * 8 + wave;
    if (row >= BATCH) return;
    const int* qr = q + (size_t)row * SHOTS;
    int s = 0;
    for (int i = lane; i < SHOTS; i += 32) s += qr[i];
#pragma unroll
    for (int off = 16; off > 0; off >>= 1) s += __shfl_xor(s, off, 32);
    if (lane == 0) e[row] = (float)s * (1.0f / (SHOTS * 15.0f));
}

// Split one float4 into bf16 hi (RNE) and bf16 lo (residual), store 8B each.
__device__ __forceinline__ void split_store4(const float4 v,
                                             __bf16* __restrict__ ph,
                                             __bf16* __restrict__ pl) {
    const float f[4] = {v.x, v.y, v.z, v.w};
    bf16x4 h, lo;
#pragma unroll
    for (int e = 0; e < 4; ++e) {
        const __bf16 hb = (__bf16)f[e];          // v_cvt_pk_bf16_f32 (paired)
        h[e]  = hb;
        lo[e] = (__bf16)(f[e] - (float)hb);      // v_fma_mix_f32_bf16 + cvt_pk
    }
    *(bf16x4*)ph = h;
    *(bf16x4*)pl = lo;
}

// 16-element bf16 fragment from two 16B LDS reads (ds_load_b128 x2).
__device__ __forceinline__ bf16x16 ld_frag16(const __bf16* p0, const __bf16* p1) {
    const bf16x8 a = *(const bf16x8*)p0;
    const bf16x8 b = *(const bf16x8*)p1;
    return __builtin_shufflevector(a, b, 0, 1, 2, 3, 4, 5, 6, 7,
                                         8, 9, 10, 11, 12, 13, 14, 15);
}

// ---------------------------------------------------------------------------
// bf16x3 WMMA GEMM + fused epilogue (bias + expectation broadcast).
// 256 threads = 8 waves, block tile 128x128, waves 4(M) x 2(N), each wave a
// 32x64 sub-tile = 2x4 accumulators -> 24 v_wmma per wave per K-chunk.
// ---------------------------------------------------------------------------
__global__ __launch_bounds__(256) void qlinear_wmma_kernel(
    const float* __restrict__ X,      // [BATCH, IN_K]
    const float* __restrict__ Wm,     // [OUT_N, IN_K]
    const float* __restrict__ bias,   // [OUT_N]
    const float* __restrict__ expv,   // [BATCH]
    float* __restrict__ out) {        // [BATCH, OUT_N]

    // Four bf16 planes, double buffered: 8 * 128*40*2B = 80 KB of 320 KB WGP LDS.
    __shared__ __bf16 lxh[2][BM * LDSH];
    __shared__ __bf16 lxl[2][BM * LDSH];
    __shared__ __bf16 lwh[2][BN * LDSH];
    __shared__ __bf16 lwl[2][BN * LDSH];

    const int t    = threadIdx.x;
    const int wave = t >> 5;
    const int lane = t & 31;
    const int wm   = wave & 3;        // wave M position: rows 32*wm .. +31
    const int wn   = wave >> 2;       // wave N position: cols 64*wn .. +63
    const int m0   = blockIdx.y * BM;
    const int n0   = blockIdx.x * BN;

    const int l  = lane & 15;         // position within half-wave
    const int uh = lane >> 4;         // 0 = lanes 0-15, 1 = lanes 16-31

    floatx8 acc[2][4];
#pragma unroll
    for (int mi = 0; mi < 2; ++mi)
#pragma unroll
        for (int ni = 0; ni < 4; ++ni) acc[mi][ni] = (floatx8)0.0f;

    // Staging map: 256 threads, 8 float4 per 32-float row; lanes 0..7 cover a
    // 128B row segment -> fully coalesced global bursts.
    const int c4 = (t & 7) * 4;       // float offset within row
    const int rr = t >> 3;            // 0..31 -> rows rr, rr+32, rr+64, rr+96

    const float* __restrict__ xrow = X  + (size_t)(m0 + rr) * IN_K + c4;
    const float* __restrict__ wrow = Wm + (size_t)(n0 + rr) * IN_K + c4;

    float4 rx[4], rw[4];

    // ---- prologue: fetch + convert + commit chunk 0 into buffer 0 ----
#pragma unroll
    for (int p = 0; p < 4; ++p) {
        rx[p] = *(const float4*)(xrow + (size_t)p * 32 * IN_K);
        rw[p] = *(const float4*)(wrow + (size_t)p * 32 * IN_K);
    }
#pragma unroll
    for (int p = 0; p < 4; ++p) {
        const int row = rr + p * 32;
        split_store4(rx[p], &lxh[0][row * LDSH + c4], &lxl[0][row * LDSH + c4]);
        split_store4(rw[p], &lwh[0][row * LDSH + c4], &lwl[0][row * LDSH + c4]);
    }
    __syncthreads();

    const int nchunks = IN_K / BK;
    for (int i = 0; i < nchunks; ++i) {
        const int  cur      = i & 1;
        const bool has_next = (i + 1) < nchunks;

        // Issue next chunk's global loads early; latency hides under WMMAs.
        if (has_next) {
            const int kb = (i + 1) * BK;
#pragma unroll
            for (int p = 0; p < 4; ++p) {
                rx[p] = *(const float4*)(xrow + (size_t)p * 32 * IN_K + kb);
                rw[p] = *(const float4*)(wrow + (size_t)p * 32 * IN_K + kb);
            }
        }

        // ---- fragments: pure ds_load_b128, ISA 7.12.2 layouts ----
        // A (16x32 MxK): lanes 0-15 row=l, K {0..7,16..23}; lanes 16-31 K {8..15,24..31}
        // B (32x16 KxN): col n = lane%16 (== W row n); lanes 0-15 K {0..7,16..23}; lanes 16-31 K {8..15,24..31}
        bf16x16 ah[2], al[2], bh[4], bl[4];
#pragma unroll
        for (int mi = 0; mi < 2; ++mi) {
            const int ro = (wm * 32 + mi * 16 + l) * LDSH;
            ah[mi] = ld_frag16(&lxh[cur][ro + uh * 8], &lxh[cur][ro + 16 + uh * 8]);
            al[mi] = ld_frag16(&lxl[cur][ro + uh * 8], &lxl[cur][ro + 16 + uh * 8]);
        }
#pragma unroll
        for (int ni = 0; ni < 4; ++ni) {
            const int ro = (wn * 64 + ni * 16 + l) * LDSH;
            bh[ni] = ld_frag16(&lwh[cur][ro + uh * 8], &lwh[cur][ro + 16 + uh * 8]);
            bl[ni] = ld_frag16(&lwl[cur][ro + uh * 8], &lwl[cur][ro + 16 + uh * 8]);
        }

        // ---- 24 x v_wmma_f32_16x16x32_bf16 ----
#pragma unroll
        for (int mi = 0; mi < 2; ++mi) {
#pragma unroll
            for (int ni = 0; ni < 4; ++ni) {
                acc[mi][ni] = __builtin_amdgcn_wmma_f32_16x16x32_bf16(
                    false, ah[mi], false, bh[ni], (short)0, acc[mi][ni], false, false);
                acc[mi][ni] = __builtin_amdgcn_wmma_f32_16x16x32_bf16(
                    false, al[mi], false, bh[ni], (short)0, acc[mi][ni], false, false);
                acc[mi][ni] = __builtin_amdgcn_wmma_f32_16x16x32_bf16(
                    false, ah[mi], false, bl[ni], (short)0, acc[mi][ni], false, false);
            }
        }

        // Convert + commit next chunk to the alternate buffer (overlaps WMMA).
        if (has_next) {
            const int nb = cur ^ 1;
#pragma unroll
            for (int p = 0; p < 4; ++p) {
                const int row = rr + p * 32;
                split_store4(rx[p], &lxh[nb][row * LDSH + c4], &lxl[nb][row * LDSH + c4]);
                split_store4(rw[p], &lwh[nb][row * LDSH + c4], &lwl[nb][row * LDSH + c4]);
            }
        }
        __syncthreads();   // one barrier per K-chunk
    }

    // ---- epilogue: C/D layout -> VGPR j holds M=j (+8 for lanes 16-31),
    //      N = lane%16.  Fuse bias[col] + expv[row]. ----
#pragma unroll
    for (int mi = 0; mi < 2; ++mi) {
        const int base_m = m0 + wm * 32 + mi * 16 + uh * 8;
#pragma unroll
        for (int ni = 0; ni < 4; ++ni) {
            const int col  = n0 + wn * 64 + ni * 16 + l;
            const float bc = bias[col];
#pragma unroll
            for (int j = 0; j < 8; ++j) {
                const int row = base_m + j;
                out[(size_t)row * OUT_N + col] = acc[mi][ni][j] + bc + expv[row];
            }
        }
    }
}

// ---------------------------------------------------------------------------
extern "C" void kernel_launch(void* const* d_in, const int* in_sizes, int n_in,
                              void* d_out, int out_size, void* d_ws, size_t ws_size,
                              hipStream_t stream) {
    const float* x    = (const float*)d_in[0];   // [8192, 4096] f32
    const float* W    = (const float*)d_in[1];   // [4096, 4096] f32
    const float* b    = (const float*)d_in[2];   // [4096] f32
    const int*   q    = (const int*)d_in[3];     // [8192, 1000] i32
    float*       out  = (float*)d_out;           // [8192, 4096] f32
    float*       expv = (float*)d_ws;            // [8192] f32 scratch (32 KB)

    qexp_kernel<<<dim3(BATCH / 8), 256, 0, stream>>>(q, expv);
    qlinear_wmma_kernel<<<dim3(OUT_N / BN, BATCH / BM), 256, 0, stream>>>(
        x, W, b, expv, out);
}
